// Trend_26998164422864
// MI455X (gfx1250) — matrix operand order repaired and checked by reading, not compiled, not run on hardware
//
#include <hip/hip_runtime.h>
#include <hip/hip_bf16.h>
#include <math.h>

#ifndef __has_builtin
#define __has_builtin(x) 0
#endif

// Flip to 0 if the assembler rejects the async-to-LDS mnemonic.
#define USE_ASYNC_LDS 1

// ---------------------------------------------------------------------------
// Problem constants: B=16, TQ=512, TKV=1024, COV=512, E=64, O=8
// ---------------------------------------------------------------------------

typedef __bf16 bf16_t;
typedef bf16_t v16bf __attribute__((ext_vector_type(16)));
typedef bf16_t v2bf  __attribute__((ext_vector_type(2)));
typedef float  v8f   __attribute__((ext_vector_type(8)));
typedef unsigned short u16;
typedef u16 v16u __attribute__((ext_vector_type(16)));
typedef u16 v8u  __attribute__((ext_vector_type(8)));

__device__ __forceinline__ u16 f32_to_bf16(float x) {
  unsigned u = __float_as_uint(x);
  u += 0x7FFFu + ((u >> 16) & 1u);          // round-to-nearest-even
  return (u16)(u >> 16);
}

// Packed conversion: two f32 -> one dword of two bf16 (v_cvt_pk_bf16_f32 if
// the toolchain exposes it, manual RNE pack otherwise).
__device__ __forceinline__ unsigned pack_bf16x2(float lo, float hi) {
#if __has_builtin(__builtin_amdgcn_cvt_pk_bf16_f32)
  v2bf r = __builtin_amdgcn_cvt_pk_bf16_f32(lo, hi);
  return __builtin_bit_cast(unsigned, r);
#else
  return (unsigned)f32_to_bf16(lo) | ((unsigned)f32_to_bf16(hi) << 16);
#endif
}

__device__ __forceinline__ void wait_async0() {
#if __has_builtin(__builtin_amdgcn_s_wait_asynccnt)
  __builtin_amdgcn_s_wait_asynccnt(0);
#else
  asm volatile("s_wait_asynccnt 0x0" ::: "memory");
#endif
}

#if USE_ASYNC_LDS
// GLOBAL_LOAD_ASYNC_TO_LDS_B128: per-lane 16B global -> 16B LDS, ASYNCcnt.
__device__ __forceinline__ void async_copy_b128(const void* gptr, void* lptr) {
  unsigned           lds = (unsigned)(size_t)lptr;       // low 32b == LDS offset
  unsigned long long ga  = (unsigned long long)(size_t)gptr;
  asm volatile("global_load_async_to_lds_b128 %0, %1, off"
               :: "v"(lds), "v"(ga) : "memory");
}
#endif

__device__ __forceinline__ v8f v8f_zero() {
  v8f z = {0.f, 0.f, 0.f, 0.f, 0.f, 0.f, 0.f, 0.f};
  return z;
}

// A-fragment (16x32 bf16, MxK): lane m(0-15) holds K = hi*8.. and 16+hi*8..
__device__ __forceinline__ v16bf frag_a(const u16* p_lo, const u16* p_hi) {
  v8u lo = *(const v8u*)p_lo;
  v8u hi = *(const v8u*)p_hi;
  v16u c;
#pragma unroll
  for (int i = 0; i < 8; ++i) { c[i] = lo[i]; c[i + 8] = hi[i]; }
  return __builtin_bit_cast(v16bf, c);
}

// B-fragment (32x16 bf16, KxN): lane (hi*16+n) holds column n, K contiguous.
__device__ __forceinline__ v16bf frag_b(const u16* p) {
  v16u c = *(const v16u*)p;
  return __builtin_bit_cast(v16bf, c);
}

__device__ __forceinline__ v8f wmma_bf16(v16bf a, v16bf b, v8f c) {
  return __builtin_amdgcn_wmma_f32_16x16x32_bf16(false, a, false, b,
                                                 (short)0, c, false, false);
}

// wave32 butterfly reductions (ds_swizzle / permlane)
__device__ __forceinline__ float wave_max(float v) {
#pragma unroll
  for (int m = 16; m > 0; m >>= 1) v = fmaxf(v, __shfl_xor(v, m, 32));
  return v;
}
__device__ __forceinline__ float wave_sum(float v) {
#pragma unroll
  for (int m = 16; m > 0; m >>= 1) v += __shfl_xor(v, m, 32);
  return v;
}

// ---------------------------------------------------------------------------
// Kernel 1: C[M,N](bf16) = A[M,512](f32) @ W[N,512](f32)^T
// 128x128 macro-tile, 8 waves, wave tile 32x64 (2x4 WMMA tiles).
// ---------------------------------------------------------------------------
__global__ __launch_bounds__(256) void k_proj(const float* __restrict__ A,
                                              const float* __restrict__ W,
                                              u16* __restrict__ C, int N) {
  __shared__ __align__(32) u16 smA[128 * 32];
  __shared__ __align__(32) u16 smB[128 * 32];
  const int tid  = threadIdx.x;
  const int lane = tid & 31;
  const int wid  = tid >> 5;
  const int nl = lane & 15, hi = lane >> 4;
  const int mBase = blockIdx.y * 128;
  const int nBase = blockIdx.x * 128;
  const int wm = wid >> 1;   // 0..3 (rows)
  const int wn = wid & 1;    // 0..1 (cols)

  v8f acc[2][4];
#pragma unroll
  for (int i = 0; i < 2; ++i)
#pragma unroll
    for (int j = 0; j < 4; ++j) acc[i][j] = v8f_zero();

  for (int k0 = 0; k0 < 512; k0 += 32) {
    // stage 128x32 slabs, f32 -> packed bf16x2 dword stores
    {
      const int base = tid * 16;        // u16 index: half a row of 32
      const int row  = base >> 5;
      const int kk   = base & 31;       // 0 or 16
      const float* pa = &A[(size_t)(mBase + row) * 512 + k0 + kk];
      const float* pw = &W[(size_t)(nBase + row) * 512 + k0 + kk];
      unsigned* da = (unsigned*)&smA[base];
      unsigned* dw = (unsigned*)&smB[base];
#pragma unroll
      for (int i = 0; i < 8; ++i) {
        da[i] = pack_bf16x2(pa[2 * i], pa[2 * i + 1]);
        dw[i] = pack_bf16x2(pw[2 * i], pw[2 * i + 1]);
      }
      if (k0 + 32 < 512) {
        __builtin_prefetch(pa + 32, 0, 1);
        __builtin_prefetch(pw + 32, 0, 1);
      }
    }
    __syncthreads();

    v16bf af[2], bfv[4];
#pragma unroll
    for (int tm = 0; tm < 2; ++tm) {
      int row = wm * 32 + tm * 16 + nl;
      af[tm] = frag_a(&smA[row * 32 + hi * 8], &smA[row * 32 + 16 + hi * 8]);
    }
#pragma unroll
    for (int tn = 0; tn < 4; ++tn) {
      int col = wn * 64 + tn * 16 + nl;
      bfv[tn] = frag_a(&smB[col * 32 + hi * 8], &smB[col * 32 + 16 + hi * 8]);
    }
#pragma unroll
    for (int tm = 0; tm < 2; ++tm)
#pragma unroll
      for (int tn = 0; tn < 4; ++tn)
        acc[tm][tn] = wmma_bf16(af[tm], bfv[tn], acc[tm][tn]);
    __syncthreads();
  }

  // C/D layout: VGPR r -> M = r + 8*hi ; N = lane&15
#pragma unroll
  for (int tm = 0; tm < 2; ++tm)
#pragma unroll
    for (int tn = 0; tn < 4; ++tn)
#pragma unroll
      for (int r = 0; r < 8; ++r) {
        int row = mBase + wm * 32 + tm * 16 + r + 8 * hi;
        int col = nBase + wn * 64 + tn * 16 + nl;
        C[(size_t)row * N + col] = f32_to_bf16(acc[tm][tn][r]);
      }
}

// ---------------------------------------------------------------------------
// Kernel 2: raw scores S = 0.125 * q @ k^T per (b,o); 128(t) x 256(s) tile.
// k-tile staged via GLOBAL_LOAD_ASYNC_TO_LDS_B128 (pure bf16 copy).
// ---------------------------------------------------------------------------
__global__ __launch_bounds__(256) void k_scores(const u16* __restrict__ qbuf,
                                                const u16* __restrict__ kvbuf,
                                                float* __restrict__ att) {
  __shared__ __align__(32) u16 smQ[128 * 64];   // 16 KB
  __shared__ __align__(32) u16 smK[256 * 64];   // 32 KB
  const int tid  = threadIdx.x;
  const int lane = tid & 31, wid = tid >> 5;
  const int nl = lane & 15, hi = lane >> 4;
  const int bo = blockIdx.z;             // b*8 + o
  const int b = bo >> 3, o = bo & 7;
  const int mBase = blockIdx.y * 128;    // t
  const int nBase = blockIdx.x * 256;    // s

  { // smK: one 128B row of k per thread
    const u16* src = &kvbuf[(size_t)(b * 1024 + nBase + tid) * 1024 + o * 64];
    u16* dst = &smK[tid * 64];
#if USE_ASYNC_LDS
#pragma unroll
    for (int c = 0; c < 8; ++c) async_copy_b128(src + c * 8, dst + c * 8);
#else
    const unsigned* s32 = (const unsigned*)src;
    unsigned* d32 = (unsigned*)dst;
#pragma unroll
    for (int i = 0; i < 32; ++i) d32[i] = s32[i];
#endif
  }
  { // smQ: 32 consecutive bf16 per thread, dword copies
    const int base = tid * 32;
    const int row = base >> 6, e0 = base & 63;
    const unsigned* src =
        (const unsigned*)&qbuf[(size_t)(b * 512 + mBase + row) * 512 + o * 64 + e0];
    unsigned* dst = (unsigned*)&smQ[base];
#pragma unroll
    for (int i = 0; i < 16; ++i) dst[i] = src[i];
  }
#if USE_ASYNC_LDS
  wait_async0();
#endif
  __syncthreads();

  const int wm = wid >> 1;   // 0..3 -> 32 rows
  const int wn = wid & 1;    // 0..1 -> 128 cols
  v8f acc[2][8];
#pragma unroll
  for (int i = 0; i < 2; ++i)
#pragma unroll
    for (int j = 0; j < 8; ++j) acc[i][j] = v8f_zero();

#pragma unroll
  for (int kc = 0; kc < 64; kc += 32) {
    v16bf af[2], bfv[8];
#pragma unroll
    for (int tm = 0; tm < 2; ++tm) {
      int row = wm * 32 + tm * 16 + nl;
      af[tm] = frag_a(&smQ[row * 64 + kc + hi * 8],
                      &smQ[row * 64 + kc + 16 + hi * 8]);
    }
#pragma unroll
    for (int tn = 0; tn < 8; ++tn) {
      int col = wn * 128 + tn * 16 + nl;
      bfv[tn] = frag_a(&smK[col * 64 + kc + hi * 8], &smK[col * 64 + kc + 16 + hi * 8]);
    }
#pragma unroll
    for (int tm = 0; tm < 2; ++tm)
#pragma unroll
      for (int tn = 0; tn < 8; ++tn)
        acc[tm][tn] = wmma_bf16(af[tm], bfv[tn], acc[tm][tn]);
  }

  const float scale = 0.125f;            // 1/sqrt(E)
#pragma unroll
  for (int tm = 0; tm < 2; ++tm)
#pragma unroll
    for (int tn = 0; tn < 8; ++tn)
#pragma unroll
      for (int r = 0; r < 8; ++r) {
        int t = mBase + wm * 32 + tm * 16 + r + 8 * hi;
        int s = nBase + wn * 128 + tn * 16 + nl;
        att[((size_t)bo * 512 + t) * 1024 + s] = scale * acc[tm][tn][r];
      }
}

// ---------------------------------------------------------------------------
// Kernel 3: row softmax over 1024 (in place) + fused att . trend_normed.
// wave32 shuffle butterflies + one 8-entry LDS combine per reduction.
// ---------------------------------------------------------------------------
__global__ __launch_bounds__(256) void k_softmax(float* __restrict__ att,
                                                 const float* __restrict__ tnorm,
                                                 float* __restrict__ tnbuf) {
  __shared__ float red[8];
  const int tid = threadIdx.x;
  const int lane = tid & 31, wid = tid >> 5;
  const size_t row = blockIdx.x;               // (b*8+o)*512 + t
  const int b = (int)(row >> 12);
  float* p = att + row * 1024;
  const int base = tid * 4;

  float4 v = *(const float4*)(p + base);
  float x0 = v.x, x1 = v.y, x2 = v.z, x3 = v.w;

  // block max
  float mx = wave_max(fmaxf(fmaxf(x0, x1), fmaxf(x2, x3)));
  if (lane == 0) red[wid] = mx;
  __syncthreads();
  mx = red[0];
#pragma unroll
  for (int i = 1; i < 8; ++i) mx = fmaxf(mx, red[i]);
  __syncthreads();

  x0 = __expf(x0 - mx); x1 = __expf(x1 - mx);
  x2 = __expf(x2 - mx); x3 = __expf(x3 - mx);

  // block sum
  float sm = wave_sum(x0 + x1 + x2 + x3);
  if (lane == 0) red[wid] = sm;
  __syncthreads();
  sm = red[0];
#pragma unroll
  for (int i = 1; i < 8; ++i) sm += red[i];
  const float inv = 1.0f / sm;
  __syncthreads();

  x0 *= inv; x1 *= inv; x2 *= inv; x3 *= inv;
  *(float4*)(p + base) = make_float4(x0, x1, x2, x3);

  // fused att . trend_normed  ('bots,bs->bto')
  const float* tn = tnorm + (size_t)b * 1024 + base;
  float dot = wave_sum(x0 * tn[0] + x1 * tn[1] + x2 * tn[2] + x3 * tn[3]);
  if (lane == 0) red[wid] = dot;
  __syncthreads();
  if (tid == 0) {
    float d = red[0];
#pragma unroll
    for (int i = 1; i < 8; ++i) d += red[i];
    tnbuf[row] = d;
  }
}

// ---------------------------------------------------------------------------
// Kernel 4: av = att @ v per (b,o). att converted f32->bf16x2 on stage; v
// staged transposed [e][s] so B-fragments are contiguous ds reads.
// ---------------------------------------------------------------------------
__global__ __launch_bounds__(256) void k_av(const float* __restrict__ att,
                                            const u16* __restrict__ kvbuf,
                                            float* __restrict__ biasbuf) {
  __shared__ __align__(32) u16 smA[128 * 64];   // 16 KB
  __shared__ __align__(32) u16 smVt[64 * 64];   // 8 KB
  const int tid  = threadIdx.x;
  const int lane = tid & 31, wid = tid >> 5;
  const int nl = lane & 15, hi = lane >> 4;
  const int bo = blockIdx.y;
  const int b = bo >> 3, o = bo & 7;
  const int mBase = blockIdx.x * 128;

  v8f acc[4];
#pragma unroll
  for (int j = 0; j < 4; ++j) acc[j] = v8f_zero();

  for (int s0 = 0; s0 < 1024; s0 += 64) {
    { // att tile: 32 consecutive elements per thread, packed dword stores
      const int base = tid * 32;
      const int row = base >> 6, ss0 = base & 63;
      const float* src = &att[((size_t)bo * 512 + mBase + row) * 1024 + s0 + ss0];
      unsigned* dst = (unsigned*)&smA[base];
#pragma unroll
      for (int i = 0; i < 16; ++i)
        dst[i] = pack_bf16x2(src[2 * i], src[2 * i + 1]);
    }
    for (int j = tid; j < 64 * 64; j += 256) {
      int ss = j >> 6, e = j & 63;   // v[s][e] -> smVt[e][s]
      smVt[e * 64 + ss] =
          kvbuf[(size_t)(b * 1024 + s0 + ss) * 1024 + 512 + o * 64 + e];
    }
    __syncthreads();

#pragma unroll
    for (int kc = 0; kc < 64; kc += 32) {
      int row = wid * 16 + nl;
      v16bf af = frag_a(&smA[row * 64 + kc + hi * 8],
                        &smA[row * 64 + kc + 16 + hi * 8]);
#pragma unroll
      for (int tn = 0; tn < 4; ++tn) {
        v16bf bfv = frag_a(&smVt[(tn * 16 + nl) * 64 + kc + hi * 8], &smVt[(tn * 16 + nl) * 64 + kc + 16 + hi * 8]);
        acc[tn] = wmma_bf16(af, bfv, acc[tn]);
      }
    }
    __syncthreads();
  }

  // biasbuf layout [b, t, o*64 + e]
#pragma unroll
  for (int tn = 0; tn < 4; ++tn)
#pragma unroll
    for (int r = 0; r < 8; ++r) {
      int t = mBase + wid * 16 + r + 8 * hi;
      int e = tn * 16 + nl;
      biasbuf[(size_t)(b * 512 + t) * 512 + o * 64 + e] = acc[tn][r];
    }
}

// ---------------------------------------------------------------------------
// Kernel 5: trend[b,t,o] = (gelu(bias @ Wfc^T) @ Wproj^T)[o] + tn[b,o,t].
// ---------------------------------------------------------------------------
__global__ __launch_bounds__(256) void k_mlp(const float* __restrict__ biasbuf,
                                             const float* __restrict__ Wfc,
                                             const float* __restrict__ Wproj,
                                             const float* __restrict__ tnbuf,
                                             float* __restrict__ trend) {
  __shared__ float smWfc[8 * 512];
  __shared__ float smWp[64];
  for (int j = threadIdx.x; j < 8 * 512; j += 256) smWfc[j] = Wfc[j];
  if (threadIdx.x < 64) smWp[threadIdx.x] = Wproj[threadIdx.x];
  __syncthreads();

  const int id = blockIdx.x * 256 + threadIdx.x;   // b*512 + t
  const int b = id >> 9, t = id & 511;
  const float* brow = biasbuf + (size_t)id * 512;

  float h[8];
#pragma unroll
  for (int j = 0; j < 8; ++j) {
    float s = 0.f;
    for (int k = 0; k < 512; k += 4) {
      float4 bv = *(const float4*)(brow + k);
      const float* w = &smWfc[j * 512 + k];
      s += bv.x * w[0] + bv.y * w[1] + bv.z * w[2] + bv.w * w[3];
    }
    h[j] = 0.5f * s * (1.0f + erff(s * 0.70710678118654752f));   // exact gelu
  }
#pragma unroll
  for (int oo = 0; oo < 8; ++oo) {
    float s = 0.f;
#pragma unroll
    for (int j = 0; j < 8; ++j) s += h[j] * smWp[oo * 8 + j];
    trend[(size_t)id * 8 + oo] = s + tnbuf[(size_t)(b * 8 + oo) * 512 + t];
  }
}

// ---------------------------------------------------------------------------
extern "C" void kernel_launch(void* const* d_in, const int* in_sizes, int n_in,
                              void* d_out, int out_size, void* d_ws,
                              size_t ws_size, hipStream_t stream) {
  (void)in_sizes; (void)n_in; (void)out_size; (void)ws_size;
  const float* trend_q  = (const float*)d_in[0];   // [16,512,512]
  const float* trend_kv = (const float*)d_in[1];   // [16,1024,512]
  const float* tnorm    = (const float*)d_in[2];   // [16,1024,1]
  const float* Wq       = (const float*)d_in[3];   // [512,512]
  const float* Wkv      = (const float*)d_in[4];   // [1024,512]
  const float* Wfc      = (const float*)d_in[5];   // [8,512]
  const float* Wproj    = (const float*)d_in[6];   // [8,8]

  float* trend_out = (float*)d_out;                         // 16*512*8
  float* att       = (float*)d_out + (size_t)16 * 512 * 8;  // 16*8*512*1024

  char* ws = (char*)d_ws;
  u16* qbuf  = (u16*)ws;            ws += (size_t)8192  * 512  * sizeof(u16);
  u16* kvbuf = (u16*)ws;            ws += (size_t)16384 * 1024 * sizeof(u16);
  float* biasbuf = (float*)ws;      ws += (size_t)8192  * 512  * sizeof(float);
  float* tnbuf   = (float*)ws;      ws += (size_t)65536 * sizeof(float);

  k_proj<<<dim3(4, 64), 256, 0, stream>>>(trend_q, Wq, qbuf, 512);
  k_proj<<<dim3(8, 128), 256, 0, stream>>>(trend_kv, Wkv, kvbuf, 1024);
  k_scores<<<dim3(4, 4, 128), 256, 0, stream>>>(qbuf, kvbuf, att);
  k_softmax<<<dim3(65536), 256, 0, stream>>>(att, tnorm, tnbuf);
  k_av<<<dim3(4, 128), 256, 0, stream>>>(att, kvbuf, biasbuf);
  k_mlp<<<dim3(32), 256, 0, stream>>>(biasbuf, Wfc, Wproj, tnbuf, trend_out);
}
